// _HFRefDecoderLayer_19000935317692
// MI455X (gfx1250) — matrix operand order repaired and checked by reading, not compiled, not run on hardware
//
#include <hip/hip_runtime.h>
#include <cstdint>

// ---------------------------------------------------------------------------
// MLA decoder layer for MI455X (gfx1250, wave32, WMMA).
// All GEMMs + attention matmuls run on v_wmma_f32_16x16x32_bf16 (bf16 in,
// fp32 accumulate). Tiles staged to LDS via b128 global loads; WMMA operands
// are contiguous 32B LDS runs (ds_load_b128). Bounds-guarded GEMM is a
// separate template instantiation so the common case has no exec-mask dance.
// ---------------------------------------------------------------------------

#define S_LEN   2048
#define DMODEL  2048
#define NHEAD   16
#define NOPE    128
#define ROPE_D  64
#define QKD     192      // NOPE + ROPE
#define VDIM    128
#define KVR     512
#define KVA_COLS 576     // KVR + ROPE
#define INTER   8192

typedef __attribute__((ext_vector_type(16))) __bf16 bf16x16;
typedef __attribute__((ext_vector_type(8)))  __bf16 bf16x8;
typedef __attribute__((ext_vector_type(4)))  __bf16 bf16x4;
typedef __attribute__((ext_vector_type(8)))  float  f32x8;
typedef __attribute__((ext_vector_type(4)))  float  f32x4;

union bfpack { bf16x16 v; bf16x8 h[2]; };

__device__ __forceinline__ f32x8 wmma_bf16f32(bf16x16 a, bf16x16 b, f32x8 c) {
  return __builtin_amdgcn_wmma_f32_16x16x32_bf16(false, a, false, b, (short)0, c,
                                                 false, false);
}

__device__ __forceinline__ bf16x4 cvt4(f32x4 v) {
  bf16x4 r;
  r[0] = (__bf16)v[0]; r[1] = (__bf16)v[1];
  r[2] = (__bf16)v[2]; r[3] = (__bf16)v[3];
  return r;
}

// ---------------------------------------------------------------------------
// RMSNorm: one block per row. Handles strided input (for kv_a[:, :512]).
// ---------------------------------------------------------------------------
__global__ __launch_bounds__(256) void rmsnorm_kernel(
    const float* __restrict__ x, int xld,
    const float* __restrict__ w,
    float* __restrict__ out, int old_, int cols)
{
  int row = blockIdx.x;
  const float* xr = x + (size_t)row * xld;
  __shared__ float red[256];
  float ss = 0.f;
  for (int c = threadIdx.x; c < cols; c += 256) { float v = xr[c]; ss += v * v; }
  red[threadIdx.x] = ss;
  __syncthreads();
  for (int st = 128; st > 0; st >>= 1) {
    if ((int)threadIdx.x < st) red[threadIdx.x] += red[threadIdx.x + st];
    __syncthreads();
  }
  float inv = rsqrtf(red[0] / (float)cols + 1e-6f);
  float* orow = out + (size_t)row * old_;
  for (int c = threadIdx.x; c < cols; c += 256) orow[c] = w[c] * (xr[c] * inv);
}

// ---------------------------------------------------------------------------
// RoPE applied in place to q_pe and k_pe. rotate_half convention.
// ---------------------------------------------------------------------------
__global__ __launch_bounds__(256) void rope_kernel(
    float* __restrict__ q, float* __restrict__ kva,
    const int* __restrict__ pos_ids)
{
  int s = blockIdx.x;
  float pos = (float)pos_ids[s];
  int t = threadIdx.x;
  const float LOGB = 9.210340371976184f; // ln(10000)
  #pragma unroll
  for (int it = 0; it < 2; ++it) {
    int p = t + it * 256;        // 0..511 : 16 heads * 32 pairs
    int hh = p >> 5, i = p & 31;
    float ang = pos * __expf(-(float)(2 * i) * (LOGB / 64.f));
    float c = __cosf(ang), sn = __sinf(ang);
    float* base = q + (size_t)s * (NHEAD * QKD) + hh * QKD + NOPE;
    float x1 = base[i], x2 = base[i + 32];
    base[i]      = x1 * c - x2 * sn;
    base[i + 32] = x2 * c + x1 * sn;
  }
  if (t < 32) {
    int i = t;
    float ang = pos * __expf(-(float)(2 * i) * (LOGB / 64.f));
    float c = __cosf(ang), sn = __sinf(ang);
    float* base = kva + (size_t)s * KVA_COLS + KVR;
    float x1 = base[i], x2 = base[i + 32];
    base[i]      = x1 * c - x2 * sn;
    base[i + 32] = x2 * c + x1 * sn;
  }
}

// ---------------------------------------------------------------------------
// Generic GEMM: C[M,N] = A[M,K] @ B[K,N] (+ Res). A,B fp32 row-major.
// Block tile 128x128, 8 waves, each 32(M)x64(N) = 2x4 WMMA tiles, BK=32.
// BOUND=true adds N-bounds checks (only the N=576 projection needs it).
// Per-thread base pointers advance by constants each K step (no muls in loop).
// ---------------------------------------------------------------------------
template <bool BOUND>
__global__ __launch_bounds__(256) void gemm_bf16_wmma(
    const float* __restrict__ A, int lda,
    const float* __restrict__ B, int ldb,
    float* __restrict__ C, int ldc,
    const float* __restrict__ Res,
    int N, int K)
{
  __shared__ __attribute__((aligned(32))) __bf16 As[128][32];   // [m][k] 8 KB
  __shared__ __attribute__((aligned(32))) __bf16 BsT[128][32];  // [n][k] 8 KB
  int tid  = threadIdx.x;
  int wave = tid >> 5, lane = tid & 31;
  int hl = lane >> 4, lr = lane & 15;
  int m0 = blockIdx.y * 128, n0 = blockIdx.x * 128;
  int wm = (wave >> 1) * 32;   // 0,32,64,96
  int wn = (wave & 1) * 64;    // 0,64
  // staging maps
  int ar = tid >> 3, ac = (tid & 7) * 4;    // A: rows ar+32i, cols ac..ac+3
  int brk = tid >> 5, bcn = (tid & 31) * 4; // B: k-rows brk+8i, n-cols bcn..+3
  const float* aP = A + (size_t)(m0 + ar) * lda + ac;   // += 32 per K step
  const float* bP = B + (size_t)brk * ldb + n0 + bcn;   // += 32*ldb per K step
  const size_t aRow = (size_t)32 * lda;                 // i*32 rows
  const size_t bRow = (size_t)8 * ldb;                  // i*8 k-rows
  bool bok = true;
  if (BOUND) bok = (n0 + bcn) < N;
  f32x8 acc[2][4] = {};

  for (int kb = 0; kb < K; kb += 32) {
    f32x4 a4[4], b4[4];
    #pragma unroll
    for (int i = 0; i < 4; ++i)
      a4[i] = *(const f32x4*)(aP + (size_t)i * aRow);
    #pragma unroll
    for (int i = 0; i < 4; ++i) {
      if (!BOUND || bok) b4[i] = *(const f32x4*)(bP + (size_t)i * bRow);
      else               b4[i] = (f32x4){0.f, 0.f, 0.f, 0.f};
    }
    if (kb + 32 < K) {  // L2 prefetch of next K tile
      __builtin_prefetch(aP + 32, 0, 1);
      if (!BOUND || bok) __builtin_prefetch(bP + (size_t)32 * ldb, 0, 1);
    }
    __syncthreads();   // previous iteration's LDS reads done
    #pragma unroll
    for (int i = 0; i < 4; ++i)
      *(bf16x4*)&As[ar + i * 32][ac] = cvt4(a4[i]);
    #pragma unroll
    for (int i = 0; i < 4; ++i) {
      bf16x4 v = cvt4(b4[i]);
      #pragma unroll
      for (int j = 0; j < 4; ++j) BsT[bcn + j][brk + i * 8] = v[j];
    }
    __syncthreads();

    bf16x16 af[2], bfv[4];
    #pragma unroll
    for (int mt = 0; mt < 2; ++mt) {      // A 16x32: k = e + 16*hl
      int row = wm + mt * 16 + lr;
      bfpack u;
      u.h[0] = *(const bf16x8*)&As[row][16 * hl];
      u.h[1] = *(const bf16x8*)&As[row][16 * hl + 8];
      af[mt] = u.v;
    }
    #pragma unroll
    for (int nt = 0; nt < 4; ++nt)        // B 32x16: k = e + 16*hl, n = lr
      bfv[nt] = *(const bf16x16*)&BsT[wn + nt * 16 + lr][16 * hl];
    #pragma unroll
    for (int mt = 0; mt < 2; ++mt)
      #pragma unroll
      for (int nt = 0; nt < 4; ++nt)
        acc[mt][nt] = wmma_bf16f32(af[mt], bfv[nt], acc[mt][nt]);

    aP += 32;
    bP += (size_t)32 * ldb;
  }

  #pragma unroll
  for (int mt = 0; mt < 2; ++mt)
    #pragma unroll
    for (int nt = 0; nt < 4; ++nt)
      #pragma unroll
      for (int r = 0; r < 8; ++r) {       // C layout: row=r+8*hl, col=lr
        int row = m0 + wm + mt * 16 + r + 8 * hl;
        int col = n0 + wn + nt * 16 + lr;
        if (!BOUND || col < N) {
          float v = acc[mt][nt][r];
          if (Res) v += Res[(size_t)row * ldc + col];
          C[(size_t)row * ldc + col] = v;
        }
      }
}

// ---------------------------------------------------------------------------
// Flash attention, causal. Block = (head, 128 query rows), 8 waves x 16 rows.
// q: [S][H*192] (rope applied), kva: [S][576] (k_pe in cols 512..),
// kvb: [S][H*256] (k_nope | v), out: [S][H*128].
// Kt[key][d] and VtT[vd][key] so both B fragments are contiguous.
// ---------------------------------------------------------------------------
__global__ __launch_bounds__(256) void attention_kernel(
    const float* __restrict__ q,
    const float* __restrict__ kva,
    const float* __restrict__ kvb,
    float* __restrict__ out)
{
  __shared__ __attribute__((aligned(32))) __bf16 Kt[32][QKD];     // 12 KB
  __shared__ __attribute__((aligned(32))) __bf16 VtT[VDIM][32];   //  8 KB
  __shared__ __attribute__((aligned(32))) __bf16 Pst[8][16][32];  //  8 KB

  int tid = threadIdx.x;
  int wave = tid >> 5, lane = tid & 31;
  int hl = lane >> 4, lr = lane & 15;
  int q0 = blockIdx.x * 128;
  int h  = blockIdx.y;
  int qbase = q0 + wave * 16;
  const int KVB_LD = NHEAD * (NOPE + VDIM);   // 4096

  // per-thread staging coordinates (fixed); pointers advance by 32 keys/iter
  int knKey = tid >> 5, knC = (tid & 31) * 4;   // nope/V: keys knKey+8*it
  int krKey = tid >> 4, krC = (tid & 15) * 4;   // rope  : keys krKey+16*it
  const float* pKn = kvb + (size_t)knKey * KVB_LD + h * (NOPE + VDIM) + knC;
  const float* pV  = pKn + NOPE;
  const float* pKr = kva + (size_t)krKey * KVA_COLS + KVR + krC;
  const size_t knStep = (size_t)8 * KVB_LD;     // +8 keys between unroll its
  const size_t krStep = (size_t)16 * KVA_COLS;  // +16 keys between unroll its

  // Q fragments (A layout), 6 chunks of K=32 covering 192 dims
  bf16x16 qf[6];
  {
    int qr = qbase + lr;
    const float* qrow = q + (size_t)qr * (NHEAD * QKD) + h * QKD;
    #pragma unroll
    for (int c = 0; c < 6; ++c)
      #pragma unroll
      for (int e = 0; e < 16; ++e)
        qf[c][e] = (__bf16)qrow[c * 32 + e + 16 * hl];
  }

  f32x8 o[8] = {};
  float mstate[8], sstate[8];
  #pragma unroll
  for (int r = 0; r < 8; ++r) { mstate[r] = -1e30f; sstate[r] = 0.f; }
  const float scale = 0.07216878364870323f;   // 1/sqrt(192)

  int nkb = q0 / 32 + 4;                      // keys 0 .. q0+127
  for (int kb = 0; kb < nkb; ++kb) {
    int k0 = kb * 32;
    __syncthreads();
    // K tile nope part: 32 keys x 128 dims
    #pragma unroll
    for (int it = 0; it < 4; ++it) {
      f32x4 v = *(const f32x4*)(pKn + it * knStep);
      *(bf16x4*)&Kt[knKey + it * 8][knC] = cvt4(v);
    }
    // K tile rope part: 32 keys x 64 dims (shared across heads)
    #pragma unroll
    for (int it = 0; it < 2; ++it) {
      f32x4 v = *(const f32x4*)(pKr + it * krStep);
      *(bf16x4*)&Kt[krKey + it * 16][NOPE + krC] = cvt4(v);
    }
    // V tile transposed: [vd][key]
    #pragma unroll
    for (int it = 0; it < 4; ++it) {
      f32x4 v = *(const f32x4*)(pV + it * knStep);
      bf16x4 b = cvt4(v);
      #pragma unroll
      for (int j = 0; j < 4; ++j) VtT[knC + j][knKey + it * 8] = b[j];
    }
    __syncthreads();

    // S = Q K^T : two 16x16 tiles (32 keys)
    f32x8 sc[2] = {};
    #pragma unroll
    for (int c = 0; c < 6; ++c) {
      #pragma unroll
      for (int nt = 0; nt < 2; ++nt) {      // B: k=e+16*hl, n=key=lr
        bf16x16 kf = *(const bf16x16*)&Kt[nt * 16 + lr][c * 32 + 16 * hl];
        sc[nt] = wmma_bf16f32(qf[c], kf, sc[nt]);
      }
    }

    // causal mask + online softmax (rows live in 16-lane halves)
    #pragma unroll
    for (int r = 0; r < 8; ++r) {
      int rowq = qbase + r + 8 * hl;
      float v0 = sc[0][r] * scale;
      float v1 = sc[1][r] * scale;
      if (k0 + lr > rowq)      v0 = -1e30f;
      if (k0 + 16 + lr > rowq) v1 = -1e30f;
      float mloc = fmaxf(v0, v1);
      #pragma unroll
      for (int off = 8; off > 0; off >>= 1)
        mloc = fmaxf(mloc, __shfl_xor(mloc, off, 32));
      float mnew = fmaxf(mstate[r], mloc);
      float corr = __expf(mstate[r] - mnew);
      float p0 = __expf(v0 - mnew);
      float p1 = __expf(v1 - mnew);
      float psum = p0 + p1;
      #pragma unroll
      for (int off = 8; off > 0; off >>= 1)
        psum += __shfl_xor(psum, off, 32);
      sstate[r] = sstate[r] * corr + psum;
      mstate[r] = mnew;
      #pragma unroll
      for (int t = 0; t < 8; ++t) o[t][r] *= corr;
      sc[0][r] = p0; sc[1][r] = p1;
    }

    // restage P (C layout -> A layout) through per-wave LDS region
    #pragma unroll
    for (int nt = 0; nt < 2; ++nt)
      #pragma unroll
      for (int r = 0; r < 8; ++r)
        Pst[wave][r + 8 * hl][nt * 16 + lr] = (__bf16)sc[nt][r];
    asm volatile("s_wait_dscnt 0" ::: "memory");
    bfpack up;
    up.h[0] = *(const bf16x8*)&Pst[wave][lr][16 * hl];
    up.h[1] = *(const bf16x8*)&Pst[wave][lr][16 * hl + 8];
    bf16x16 pf = up.v;

    // O += P V  (8 tiles over VD=128)
    #pragma unroll
    for (int t = 0; t < 8; ++t) {
      bf16x16 vf = *(const bf16x16*)&VtT[t * 16 + lr][16 * hl];
      o[t] = wmma_bf16f32(pf, vf, o[t]);
    }

    pKn += (size_t)32 * KVB_LD;
    pV  += (size_t)32 * KVB_LD;
    pKr += (size_t)32 * KVA_COLS;
  }

  #pragma unroll
  for (int t = 0; t < 8; ++t)
    #pragma unroll
    for (int r = 0; r < 8; ++r) {
      int row = qbase + r + 8 * hl;
      out[(size_t)row * (NHEAD * VDIM) + h * VDIM + t * 16 + lr] =
          o[t][r] / sstate[r];
    }
}

// ---------------------------------------------------------------------------
// act = silu(gate) * up, in place into gate
// ---------------------------------------------------------------------------
__global__ __launch_bounds__(256) void silu_mul_kernel(
    float* __restrict__ g, const float* __restrict__ u, long long n)
{
  long long i = (long long)blockIdx.x * 256 + threadIdx.x;
  long long stride = (long long)gridDim.x * 256;
  for (; i < n; i += stride) {
    float x = g[i];
    float s = x / (1.f + __expf(-x));
    g[i] = s * u[i];
  }
}

// ---------------------------------------------------------------------------
extern "C" void kernel_launch(void* const* d_in, const int* in_sizes, int n_in,
                              void* d_out, int out_size, void* d_ws, size_t ws_size,
                              hipStream_t stream) {
  (void)in_sizes; (void)n_in; (void)out_size; (void)ws_size;
  const float* hidden = (const float*)d_in[0];
  const int*   pos    = (const int*)d_in[1];
  const float* ln1w   = (const float*)d_in[2];
  const float* ln2w   = (const float*)d_in[3];
  const float* Wq     = (const float*)d_in[4];
  const float* Wkva   = (const float*)d_in[5];
  const float* kvlnw  = (const float*)d_in[6];
  const float* Wkvb   = (const float*)d_in[7];
  const float* Wo     = (const float*)d_in[8];
  const float* Wgate  = (const float*)d_in[9];
  const float* Wup    = (const float*)d_in[10];
  const float* Wdown  = (const float*)d_in[11];
  float* outp = (float*)d_out;

  float* ws = (float*)d_ws;
  size_t off = 0;
  float* hsn  = ws + off; off += (size_t)S_LEN * DMODEL;              // ln1/ln2 out
  float* qbuf = ws + off; off += (size_t)S_LEN * NHEAD * QKD;         // q
  float* kva  = ws + off; off += (size_t)S_LEN * KVA_COLS;            // kv_a
  float* ckv  = ws + off; off += (size_t)S_LEN * KVR;                 // normed ckv
  float* kvb  = ws + off; off += (size_t)S_LEN * NHEAD * (NOPE+VDIM); // kv_b
  float* attn = ws + off; off += (size_t)S_LEN * NHEAD * VDIM;        // attn out
  float* hs2  = ws + off; off += (size_t)S_LEN * DMODEL;              // mid residual
  float* gate = ws + off; off += (size_t)S_LEN * INTER;               // gate/act
  float* up   = ws + off; off += (size_t)S_LEN * INTER;               // up

  dim3 blk(256);

  rmsnorm_kernel<<<S_LEN, blk, 0, stream>>>(hidden, DMODEL, ln1w, hsn, DMODEL, DMODEL);

  gemm_bf16_wmma<false><<<dim3((NHEAD*QKD)/128, S_LEN/128), blk, 0, stream>>>(
      hsn, DMODEL, Wq, NHEAD*QKD, qbuf, NHEAD*QKD, nullptr, NHEAD*QKD, DMODEL);

  gemm_bf16_wmma<true><<<dim3((KVA_COLS + 127)/128, S_LEN/128), blk, 0, stream>>>(
      hsn, DMODEL, Wkva, KVA_COLS, kva, KVA_COLS, nullptr, KVA_COLS, DMODEL);

  rmsnorm_kernel<<<S_LEN, blk, 0, stream>>>(kva, KVA_COLS, kvlnw, ckv, KVR, KVR);

  rope_kernel<<<S_LEN, blk, 0, stream>>>(qbuf, kva, pos);

  gemm_bf16_wmma<false><<<dim3((NHEAD*(NOPE+VDIM))/128, S_LEN/128), blk, 0, stream>>>(
      ckv, KVR, Wkvb, NHEAD*(NOPE+VDIM), kvb, NHEAD*(NOPE+VDIM), nullptr,
      NHEAD*(NOPE+VDIM), KVR);

  attention_kernel<<<dim3(S_LEN/128, NHEAD), blk, 0, stream>>>(qbuf, kva, kvb, attn);

  gemm_bf16_wmma<false><<<dim3(DMODEL/128, S_LEN/128), blk, 0, stream>>>(
      attn, NHEAD*VDIM, Wo, DMODEL, hs2, DMODEL, hidden, DMODEL, NHEAD*VDIM);

  rmsnorm_kernel<<<S_LEN, blk, 0, stream>>>(hs2, DMODEL, ln2w, hsn, DMODEL, DMODEL);

  gemm_bf16_wmma<false><<<dim3(INTER/128, S_LEN/128), blk, 0, stream>>>(
      hsn, DMODEL, Wgate, INTER, gate, INTER, nullptr, INTER, DMODEL);

  gemm_bf16_wmma<false><<<dim3(INTER/128, S_LEN/128), blk, 0, stream>>>(
      hsn, DMODEL, Wup, INTER, up, INTER, nullptr, INTER, DMODEL);

  silu_mul_kernel<<<dim3(2048), blk, 0, stream>>>(gate, up, (long long)S_LEN * INTER);

  // C = hs2 + act @ Wdown : A=gate[2048,8192], B=Wdown[8192,2048]
  gemm_bf16_wmma<false><<<dim3(DMODEL/128, S_LEN/128), blk, 0, stream>>>(
      gate, INTER, Wdown, DMODEL, outp, DMODEL, hs2, DMODEL, INTER);
}
